// PointNetSetAbstractionMSG_1022202217391
// MI455X (gfx1250) — matrix operand order repaired and checked by reading, not compiled, not run on hardware
//
#include <hip/hip_runtime.h>

typedef __attribute__((ext_vector_type(16))) _Float16 v16h;
typedef __attribute__((ext_vector_type(8)))  _Float16 v8h;
typedef __attribute__((ext_vector_type(8)))  float    v8f;

constexpr int BATCH = 8;
constexpr int NPTS  = 16384;
constexpr int NQ    = 1024;   // NPOINT
constexpr int CTOT  = 320;    // 64 + 128 + 128
#define BIGF 1.0e10f

// ---------------------------------------------------------------------------
// helpers
// ---------------------------------------------------------------------------
static __device__ __forceinline__ unsigned long long shfl_xor_u64(unsigned long long v, int m) {
  unsigned int lo = (unsigned int)v, hi = (unsigned int)(v >> 32);
  lo = __shfl_xor(lo, m, 32);
  hi = __shfl_xor(hi, m, 32);
  return ((unsigned long long)hi << 32) | lo;
}

// block argmax over (val desc, idx asc) — exact jnp.argmax tie-break.
// requires blockDim.x == 1024 (32 waves).
static __device__ __forceinline__ int block_argmax(float bv, int bi,
                                                   float* swv, int* swi, int* sres,
                                                   int lane, int wid) {
  #pragma unroll
  for (int off = 16; off; off >>= 1) {
    float ov = __shfl_xor(bv, off, 32);
    int   oi = __shfl_xor(bi, off, 32);
    if (ov > bv || (ov == bv && oi < bi)) { bv = ov; bi = oi; }
  }
  if (lane == 0) { swv[wid] = bv; swi[wid] = bi; }
  __syncthreads();
  if (wid == 0) {
    bv = swv[lane]; bi = swi[lane];
    #pragma unroll
    for (int off = 16; off; off >>= 1) {
      float ov = __shfl_xor(bv, off, 32);
      int   oi = __shfl_xor(bi, off, 32);
      if (ov > bv || (ov == bv && oi < bi)) { bv = ov; bi = oi; }
    }
    if (lane == 0) *sres = bi;
  }
  __syncthreads();
  return *sres;
}

// ---------------------------------------------------------------------------
// 1) Farthest point sampling (masked, reference-faithful) + mask popcount
//    grid = BATCH, block = 1024. Writes new_xyz directly.
// ---------------------------------------------------------------------------
__global__ __launch_bounds__(1024) void fps_kernel(const float* __restrict__ xyz,
                                                   const int*   __restrict__ mask,
                                                   float*       __restrict__ newxyz,
                                                   int*         __restrict__ counts) {
  const int b = blockIdx.x, tid = threadIdx.x;
  const int lane = tid & 31, wid = tid >> 5;
  __shared__ float swv[32];
  __shared__ int   swi[32];
  __shared__ int   sbi;

  float px[16], py[16], pz[16], dist[16];
  unsigned int mbits = 0;
  #pragma unroll
  for (int j = 0; j < 16; ++j) {
    int i = tid + 1024 * j;
    const float* p = xyz + ((size_t)b * NPTS + i) * 3;
    px[j] = p[0]; py[j] = p[1]; pz[j] = p[2];
    int m = mask[(size_t)b * NPTS + i];
    if (m) mbits |= 1u << j;
    dist[j] = m ? 0.0f : BIGF;   // reference: where(mask, 0, BIG)
  }

  // valid_counts = sum(mask)
  int cnt = __popc(mbits);
  #pragma unroll
  for (int off = 16; off; off >>= 1) cnt += __shfl_xor(cnt, off, 32);
  if (lane == 0) swi[wid] = cnt;
  __syncthreads();
  if (tid == 0) { int c = 0; for (int w = 0; w < 32; ++w) c += swi[w]; counts[b] = c; }
  __syncthreads();

  // farthest = argmax(mask)
  float bv = -1.f; int bi = 0x7fffffff;
  #pragma unroll
  for (int j = 0; j < 16; ++j) {
    float v = ((mbits >> j) & 1u) ? 1.f : 0.f;
    int i = tid + 1024 * j;
    if (v > bv || (v == bv && i < bi)) { bv = v; bi = i; }
  }
  int far = block_argmax(bv, bi, swv, swi, &sbi, lane, wid);

  for (int it = 0; it < NQ; ++it) {
    const float* c = xyz + ((size_t)b * NPTS + far) * 3;
    float cx = c[0], cy = c[1], cz = c[2];
    if (tid == 0) {
      float* o = newxyz + ((size_t)b * NQ + it) * 3;
      o[0] = cx; o[1] = cy; o[2] = cz;       // new_xyz = xyz[cents]
    }
    float lv = -1.f; int li = 0x7fffffff;
    #pragma unroll
    for (int j = 0; j < 16; ++j) {
      float dx = px[j] - cx, dy = py[j] - cy, dz = pz[j] - cz;
      float d = dx * dx + dy * dy + dz * dz;
      bool m = (mbits >> j) & 1u;
      if (d < dist[j] && m) dist[j] = d;     // upd = (d < dist) & mask
      int i = tid + 1024 * j;
      if (dist[j] > lv || (dist[j] == lv && i < li)) { lv = dist[j]; li = i; }
    }
    far = block_argmax(lv, li, swv, swi, &sbi, lane, wid);
  }
}

// ---------------------------------------------------------------------------
// 2) top-64 nearest neighbors per query (prefix gives top-16/32).
//    key = (ordered_bits(dist)<<32)|idx  ->  min == top_k(-dist) order exactly.
//    grid = (NQ, BATCH), block = 512.
// ---------------------------------------------------------------------------
__global__ __launch_bounds__(512) void knn_kernel(const float* __restrict__ xyz,
                                                  const int*   __restrict__ mask,
                                                  const float* __restrict__ newxyz,
                                                  int*         __restrict__ knn) {
  const int s = blockIdx.x, b = blockIdx.y, tid = threadIdx.x;
  const int lane = tid & 31, wid = tid >> 5;
  __shared__ unsigned long long swk[16];
  __shared__ unsigned long long swin;

  const float* q = newxyz + ((size_t)b * NQ + s) * 3;
  float sx = q[0], sy = q[1], sz = q[2];
  float s2 = sx * sx + sy * sy + sz * sz;

  unsigned long long key[32];
  #pragma unroll
  for (int j = 0; j < 32; ++j) {
    int i = tid + 512 * j;
    const float* p = xyz + ((size_t)b * NPTS + i) * 3;
    float x = p[0], y = p[1], z = p[2];
    float p2 = x * x + y * y + z * z;
    float dot = sx * x + sy * y + sz * z;
    float d = -2.f * dot + s2 + p2;          // reference square_distance
    if (!mask[(size_t)b * NPTS + i]) d = BIGF;
    unsigned int u = __float_as_uint(d);
    u = (u & 0x80000000u) ? ~u : (u | 0x80000000u);   // total order incl. negatives
    key[j] = ((unsigned long long)u << 32) | (unsigned int)i;
  }

  int* outq = knn + ((size_t)b * NQ + s) * 64;
  for (int r = 0; r < 64; ++r) {
    unsigned long long k = ~0ull;
    #pragma unroll
    for (int j = 0; j < 32; ++j) k = key[j] < k ? key[j] : k;
    #pragma unroll
    for (int off = 16; off; off >>= 1) {
      unsigned long long o = shfl_xor_u64(k, off);
      if (o < k) k = o;
    }
    if (lane == 0) swk[wid] = k;
    __syncthreads();
    if (wid == 0) {
      unsigned long long v = (lane < 16) ? swk[lane] : ~0ull;
      #pragma unroll
      for (int off = 16; off; off >>= 1) {
        unsigned long long o = shfl_xor_u64(v, off);
        if (o < v) v = o;
      }
      if (lane == 0) swin = v;
    }
    __syncthreads();
    unsigned long long w = swin;
    #pragma unroll
    for (int j = 0; j < 32; ++j) if (key[j] == w) key[j] = ~0ull;
    if (tid == 0) outq[r] = (int)(w & 0xffffffffu);
  }
}

// ---------------------------------------------------------------------------
// 3) gather + fused 3-layer pointwise MLP (WMMA f16->f32) + max-pool.
//    Templated per scale: all divisions/strides/trip-counts fold to constants,
//    tile and K loops fully unroll around v_wmma_f32_16x16x32_f16.
//    block = 256 threads (8 waves), 64 rows x full channel width per block.
// ---------------------------------------------------------------------------
template <int NS, int C1, int C2, int C3, int CHOFF>
__global__ __launch_bounds__(256) void mlp_kernel(
    const float* __restrict__ xyz, const float* __restrict__ points,
    const int* __restrict__ knn, const int* __restrict__ counts,
    const float* __restrict__ newxyz,
    const float* __restrict__ W0p, const float* __restrict__ B0p,
    const float* __restrict__ W1p, const float* __restrict__ B1p,
    const float* __restrict__ W2p, const float* __restrict__ B2p,
    float* __restrict__ out) {
  constexpr int AS = 136;   // act row stride (halves): 272B -> conflict-free
  constexpr int WS = 136;   // W^T row stride (halves)
  constexpr int FS = 132;   // final f32 row stride (floats)
  constexpr int TM = 64;    // rows per block

  __shared__ __align__(16) _Float16      sA[TM * AS];         // 17 KB
  __shared__ __align__(16) unsigned char sPool[TM * FS * 4];  // 33 KB: actB(f16) / final(f32)
  __shared__ __align__(16) _Float16      sW[128 * WS];        // 34 KB
  __shared__ float sBias[128];

  _Float16* sB   = (_Float16*)sPool;
  float*    sFin = (float*)sPool;

  const int tid = threadIdx.x, lane = tid & 31, wave = tid >> 5;

  __builtin_prefetch(W0p, 0, 2);
  __builtin_prefetch(W1p, 0, 2);
  __builtin_prefetch(W2p, 0, 2);

  // ---- gather stage: build layer-0 input [TM x 32], channels 9..31 zero ----
  if (tid < TM) {
    int R  = blockIdx.x * TM + tid;          // global row = (b*NQ+s)*NS + k
    int k  = R % NS;                         // NS pow2 -> and
    int sf = R / NS;                         // b*NQ + s
    int b  = sf / NQ;
    int j  = (counts[b] < NS) ? 0 : k;       // overflow -> idx[...,0]
    int g  = knn[(size_t)sf * 64 + j];
    const float* c = newxyz + (size_t)sf * 3;
    const float* p = xyz    + ((size_t)b * NPTS + g) * 3;
    const float* f = points + ((size_t)b * NPTS + g) * 6;
    _Float16* row = sA + tid * AS;
    row[0] = (_Float16)(p[0] - c[0]);
    row[1] = (_Float16)(p[1] - c[1]);
    row[2] = (_Float16)(p[2] - c[2]);
    #pragma unroll
    for (int q = 0; q < 6; ++q) row[3 + q] = (_Float16)f[q];
    #pragma unroll
    for (int q = 9; q < 32; ++q) row[q] = (_Float16)0.f;
  }

  const float* Wl[3] = {W0p, W1p, W2p};
  const float* Bl[3] = {B0p, B1p, B2p};
  constexpr int CO[3]  = {C1, C2, C3};
  constexpr int KIN[3] = {9, C1, C2};
  constexpr int KPA[3] = {32, C1, C2};   // C1,C2 are multiples of 32

  #pragma unroll
  for (int l = 0; l < 3; ++l) {
    __syncthreads();           // prior stage complete before overwriting sW
    const float* W  = Wl[l];
    const float* Bb = Bl[l];
    const int Cout = CO[l];
    const int KP   = KPA[l];
    const int Kin  = KIN[l];
    const _Float16* inAct = (l == 1) ? sB : sA;
    _Float16*       oAct  = (l == 0) ? sB : sA;

    // stage W^T (f16, zero-padded K) + bias.
    // Coalesced: thread -> float4 of 4 consecutive n at fixed kk (contiguous,
    // 16B-aligned since Cout % 4 == 0); scatter 4 f16 into transposed LDS.
    {
      const int CV = Cout >> 2;            // float4 groups per W row
      #pragma unroll 2
      for (int idx = tid; idx < KP * CV; idx += 256) {
        int kk = idx / CV;                 // const divisor -> mul-shift
        int nq = idx - kk * CV;
        float4 wv = make_float4(0.f, 0.f, 0.f, 0.f);
        if (kk < Kin) wv = *(const float4*)(W + (size_t)kk * Cout + 4 * nq);
        sW[(4 * nq + 0) * WS + kk] = (_Float16)wv.x;
        sW[(4 * nq + 1) * WS + kk] = (_Float16)wv.y;
        sW[(4 * nq + 2) * WS + kk] = (_Float16)wv.z;
        sW[(4 * nq + 3) * WS + kk] = (_Float16)wv.w;
      }
    }
    if (tid < Cout) sBias[tid] = Bb[tid];
    __syncthreads();

    const int NT    = Cout >> 4;        // tiles along N
    const int TILES = (TM / 16) * NT;   // 8,16,24,32 — always multiple of 8
    #pragma unroll
    for (int i = 0; i < TILES / 8; ++i) {
      const int t  = wave + i * 8;      // nt constant per wave when NT|8
      const int mt = t / NT, nt = t - mt * NT;
      v8f acc = {0.f, 0.f, 0.f, 0.f, 0.f, 0.f, 0.f, 0.f};
      #pragma unroll
      for (int kbi = 0; kbi < 4; ++kbi) {
        if (kbi * 32 >= KP) break;      // compile-time prune after unroll
        const int kb = kbi * 32;
        // A fragment: 16x32 f16, ISA lane layout
        int arow = mt * 16 + (lane & 15);
        int ak   = kb + ((lane & 16) ? 8 : 0);
        const _Float16* ap = inAct + arow * AS + ak;
        v8h alo = *(const v8h*)ap;
        v8h ahi = *(const v8h*)(ap + 16);
        v16h af = __builtin_shufflevector(alo, ahi, 0,1,2,3,4,5,6,7,8,9,10,11,12,13,14,15);
        // B fragment: 32x16 f16, same K layout as A (8 halves, then 8 halves at +16)
        int bcol = nt * 16 + (lane & 15);
        int bk   = kb + ((lane & 16) ? 8 : 0);
        const _Float16* bp = sW + bcol * WS + bk;
        v8h blo = *(const v8h*)bp;
        v8h bhi = *(const v8h*)(bp + 16);
        v16h bf = __builtin_shufflevector(blo, bhi, 0,1,2,3,4,5,6,7,8,9,10,11,12,13,14,15);
        acc = __builtin_amdgcn_wmma_f32_16x16x32_f16(false, af, false, bf,
                                                     (short)0, acc, false, false);
      }
      // D layout: lane->N, VGPR r -> M = r (+8 for hi lanes); bias + ReLU
      int col   = nt * 16 + (lane & 15);
      int rbase = mt * 16 + ((lane & 16) ? 8 : 0);
      float bv = sBias[col];
      #pragma unroll
      for (int r = 0; r < 8; ++r) {
        float v = fmaxf(acc[r] + bv, 0.f);
        int row = rbase + r;
        if (l == 2) sFin[row * FS + col] = v;
        else        oAct[row * AS + col] = (_Float16)v;
      }
    }
  }
  __syncthreads();

  // ---- max over NS rows per group, write output slice ----
  constexpr int GROUPS = TM / NS;
  for (int p = tid; p < GROUPS * C3; p += 256) {
    int g = p / C3, c = p - g * C3;
    float m = 0.f;                       // post-ReLU values are >= 0
    #pragma unroll 4
    for (int k = 0; k < NS; ++k) m = fmaxf(m, sFin[(g * NS + k) * FS + c]);
    int sf = (blockIdx.x * TM) / NS + g; // = b*NQ + s
    out[(size_t)sf * CTOT + CHOFF + c] = m;
  }
}

// ---------------------------------------------------------------------------
// launch
// ---------------------------------------------------------------------------
extern "C" void kernel_launch(void* const* d_in, const int* in_sizes, int n_in,
                              void* d_out, int out_size, void* d_ws, size_t ws_size,
                              hipStream_t stream) {
  (void)in_sizes; (void)n_in; (void)out_size; (void)ws_size;
  const float* xyz    = (const float*)d_in[0];
  const float* points = (const float*)d_in[1];
  const int*   mask   = (const int*)d_in[2];

  float* out    = (float*)d_out;
  float* newxyz = out;                              // [B,1024,3]
  float* newpts = out + (size_t)BATCH * NQ * 3;     // [B,1024,320]

  int* counts = (int*)d_ws;                         // [B]
  int* knn    = (int*)((char*)d_ws + 256);          // [B,1024,64]

  fps_kernel<<<BATCH, 1024, 0, stream>>>(xyz, mask, newxyz, counts);
  knn_kernel<<<dim3(NQ, BATCH), 512, 0, stream>>>(xyz, mask, newxyz, knn);

  const float* w[18];
  for (int i = 0; i < 18; ++i) w[i] = (const float*)d_in[3 + i];

  // scale 0: ns=16, mlp 9->32->32->64, out channels [0,64)
  mlp_kernel<16, 32, 32, 64, 0>
      <<<BATCH * NQ * 16 / 64, 256, 0, stream>>>(xyz, points, knn, counts, newxyz,
                                                 w[0], w[1], w[2], w[3], w[4], w[5], newpts);
  // scale 1: ns=32, mlp 9->64->64->128, out channels [64,192)
  mlp_kernel<32, 64, 64, 128, 64>
      <<<BATCH * NQ * 32 / 64, 256, 0, stream>>>(xyz, points, knn, counts, newxyz,
                                                 w[6], w[7], w[8], w[9], w[10], w[11], newpts);
  // scale 2: ns=64, mlp 9->64->96->128, out channels [192,320)
  mlp_kernel<64, 64, 96, 128, 192>
      <<<BATCH * NQ * 64 / 64, 256, 0, stream>>>(xyz, points, knn, counts, newxyz,
                                                 w[12], w[13], w[14], w[15], w[16], w[17], newpts);
}
